// SurpriseGatedStore_74586402063015
// MI455X (gfx1250) — matrix operand order repaired and checked by reading, not compiled, not run on hardware
//
#include <hip/hip_runtime.h>
#include <hip/hip_bf16.h>

// Problem shape (fixed by the reference)
#define BB 2
#define SS 1024
#define DD 1024
#define NN 8192
#define TT (BB * SS)   // 2048 tokens

typedef __attribute__((ext_vector_type(16))) __bf16 v16bf;
typedef __attribute__((ext_vector_type(8)))  float  v8f;

// CDNA5 async global->LDS copy path (ASYNCcnt). Probe-derived signature:
//   void __builtin_amdgcn_global_load_async_to_lds_b128(v4i* src, v4i as3* dst,
//                                                       imm int offset, imm int cpol)
#if __has_builtin(__builtin_amdgcn_global_load_async_to_lds_b128) && \
    __has_builtin(__builtin_amdgcn_s_wait_asynccnt)
#define HAVE_ASYNC_LDS 1
typedef int v4i __attribute__((vector_size(16)));
typedef __attribute__((address_space(3))) v4i lds_v4i;
#else
#define HAVE_ASYNC_LDS 0
#endif

// ---------------------------------------------------------------------------
// ctx_in[t] = (t % S == 0) ? 0 : x[t-1]   (shift-right by one token, per batch)
// ---------------------------------------------------------------------------
__global__ __launch_bounds__(256)
void shift_tokens_kernel(const float* __restrict__ x, float* __restrict__ ctxin)
{
    const int t = blockIdx.x;
    float4* dst = (float4*)(ctxin + (size_t)t * DD);
    if ((t & (SS - 1)) == 0) {
        float4 z; z.x = z.y = z.z = z.w = 0.0f;
        dst[threadIdx.x] = z;
    } else {
        const float4* src = (const float4*)(x + (size_t)(t - 1) * DD);
        dst[threadIdx.x] = src[threadIdx.x];
    }
}

// --- fragment helpers -------------------------------------------------------
__device__ __forceinline__ void cvt8(v16bf& d, int base, float4 a, float4 b)
{
    d[base + 0] = (__bf16)a.x; d[base + 1] = (__bf16)a.y;
    d[base + 2] = (__bf16)a.z; d[base + 3] = (__bf16)a.w;
    d[base + 4] = (__bf16)b.x; d[base + 5] = (__bf16)b.y;
    d[base + 6] = (__bf16)b.z; d[base + 7] = (__bf16)b.w;
}
// A fragment (16x32 bf16): lane row = rp, k = k0 + half*16 + {0..15}
__device__ __forceinline__ v16bf load_a_frag(const float* rp, int k0, int half)
{
    const float4* p0 = (const float4*)(rp + k0 + half * 16);
    const float4* p1 = (const float4*)(rp + k0 + half * 16 + 8);
    v16bf f; cvt8(f, 0, p0[0], p0[1]); cvt8(f, 8, p1[0], p1[1]); return f;
}
// B^T fragment (32x16 bf16): lane col row = rp, k = k0 + half*16 + {0..15}
__device__ __forceinline__ v16bf load_bt_frag(const float* rp, int k0, int half)
{
    const float4* p = (const float4*)(rp + k0 + half * 16);
    v16bf f; cvt8(f, 0, p[0], p[1]); cvt8(f, 8, p[2], p[3]); return f;
}
// B fragment from the LDS tile: element e holds k = e + half*16 at column col
__device__ __forceinline__ v16bf load_blds_frag(const float (*tile)[128], int col, int half)
{
    v16bf f;
#pragma unroll
    for (int e = 0; e < 16; ++e) f[e] = (__bf16)tile[e + half * 16][col];
    return f;
}

// ---------------------------------------------------------------------------
// Generic WMMA GEMM: Y[M,N] = epilogue( scale * (A[M,K] @ B) + bias )
//   bT=0: B is [K,N] row-major, staged through a 16 KB LDS tile (async copy
//         via GLOBAL_LOAD_ASYNC_TO_LDS_B128 when the toolchain exposes it).
//   bT=1: B is [N,K] row-major (A @ B^T), loaded directly (contiguous per lane).
//   act=1: exact GELU.  maskv: if maskv[col]==0 -> write -inf (slot masking).
// 256 threads = 8 waves in 2(M)x4(N); each wave computes a 32x32 output
// (2x2 register-blocked 16x16 WMMA tiles); block tile 64x128; K-step 32.
// ---------------------------------------------------------------------------
__global__ __launch_bounds__(256)
void wmma_gemm_kernel(const float* __restrict__ A, const float* __restrict__ Bm,
                      const float* __restrict__ bias, const float* __restrict__ maskv,
                      float* __restrict__ Y,
                      int M, int K, int N, int bT, int act, float scale)
{
    __shared__ float tileB[32][128];          // only used when bT==0 (16 KB)

    const int tid  = threadIdx.x;
    const int lane = tid & 31;
    const int wave = tid >> 5;                // 0..7
    const int wm   = wave >> 2;               // 0..1  (M)
    const int wn   = wave & 3;                // 0..3  (N)
    const int bm   = blockIdx.x * 64 + wm * 32;
    const int bnb  = blockIdx.y * 128;        // block column base
    const int bn   = bnb + wn * 32;
    const int half = lane >> 4;               // 0 or 1
    const int l15  = lane & 15;

    const float* arp0 = A + (size_t)(bm + l15) * K;
    const float* arp1 = A + (size_t)(bm + 16 + l15) * K;
    const float* btp0 = bT ? (Bm + (size_t)(bn + l15) * K) : nullptr;
    const float* btp1 = bT ? (Bm + (size_t)(bn + 16 + l15) * K) : nullptr;

    v8f c00 = {}, c01 = {}, c10 = {}, c11 = {};

    for (int k0 = 0; k0 < K; k0 += 32) {
        if (!bT) {
            __syncthreads();  // previous iteration's readers done with tileB
            // Stage B[k0:k0+32, bnb:bnb+128]: 1024 float4, 4 per thread, coalesced.
#pragma unroll
            for (int i = 0; i < 4; ++i) {
                const int f  = tid + i * 256;          // float4 index 0..1023
                const int kk = f >> 5;                 // 0..31
                const int c4 = (f & 31) << 2;          // 0,4,..,124
                const float* gsrc = Bm + (size_t)(k0 + kk) * N + bnb + c4;
                float* ldst = &tileB[kk][c4];
#if HAVE_ASYNC_LDS
                __builtin_amdgcn_global_load_async_to_lds_b128(
                    (v4i*)gsrc, (lds_v4i*)ldst, 0, 0);
#else
                *(float4*)ldst = *(const float4*)gsrc;
#endif
            }
#if HAVE_ASYNC_LDS
            __builtin_amdgcn_s_wait_asynccnt(0);
#endif
            __syncthreads();
        }

        __builtin_prefetch(arp0 + k0 + 128, 0, 1);   // speculative, branch-free

        const v16bf af0 = load_a_frag(arp0, k0, half);
        const v16bf af1 = load_a_frag(arp1, k0, half);

        v16bf bf0, bf1;
        if (bT) {
            __builtin_prefetch(btp0 + k0 + 128, 0, 1);
            bf0 = load_bt_frag(btp0, k0, half);
            bf1 = load_bt_frag(btp1, k0, half);
        } else {
            bf0 = load_blds_frag(tileB, wn * 32 + l15, half);
            bf1 = load_blds_frag(tileB, wn * 32 + 16 + l15, half);
        }

        c00 = __builtin_amdgcn_wmma_f32_16x16x32_bf16(false, af0, false, bf0, (short)0, c00, false, false);
        c01 = __builtin_amdgcn_wmma_f32_16x16x32_bf16(false, af0, false, bf1, (short)0, c01, false, false);
        c10 = __builtin_amdgcn_wmma_f32_16x16x32_bf16(false, af1, false, bf0, (short)0, c10, false, false);
        c11 = __builtin_amdgcn_wmma_f32_16x16x32_bf16(false, af1, false, bf1, (short)0, c11, false, false);
    }

    // Epilogue: scale, bias, activation, mask; C/D layout row = r + half*8.
    auto store_tile = [&](const v8f& c, int rbase, int cbase) {
        const int col  = cbase + l15;
        const float ba = bias ? bias[col] : 0.0f;
        const bool msk = maskv && (maskv[col] == 0.0f);
#pragma unroll
        for (int r = 0; r < 8; ++r) {
            const int row = rbase + r + half * 8;
            float v = c[r] * scale + ba;
            if (act == 1) v = 0.5f * v * (1.0f + erff(v * 0.70710678118654752f));
            if (msk) v = -__builtin_inff();
            Y[(size_t)row * N + col] = v;
        }
    };
    store_tile(c00, bm,      bn);
    store_tile(c01, bm,      bn + 16);
    store_tile(c10, bm + 16, bn);
    store_tile(c11, bm + 16, bn + 16);
}

// ---------------------------------------------------------------------------
// surprise[t] = sqrt(mean_d (x[t,d] - pred[t,d])^2)
// ---------------------------------------------------------------------------
__global__ __launch_bounds__(256)
void surprise_kernel(const float* __restrict__ x, const float* __restrict__ pred,
                     float* __restrict__ surprise)
{
    const int t = blockIdx.x;
    __shared__ float red[256];
    float acc = 0.0f;
    for (int d = threadIdx.x; d < DD; d += 256) {
        const float df = x[(size_t)t * DD + d] - pred[(size_t)t * DD + d];
        acc += df * df;
    }
    red[threadIdx.x] = acc; __syncthreads();
    for (int s = 128; s > 0; s >>= 1) {
        if (threadIdx.x < s) red[threadIdx.x] += red[threadIdx.x + s];
        __syncthreads();
    }
    if (threadIdx.x == 0) surprise[t] = sqrtf(red[0] / (float)DD);
}

// ---------------------------------------------------------------------------
// threshold = (mean + 1.0*std_unbiased) * (1 + util*0.1), std clamped to 1e-6
// ---------------------------------------------------------------------------
__global__ __launch_bounds__(1024)
void threshold_kernel(const float* __restrict__ surprise,
                      const float* __restrict__ sl0, float* __restrict__ thr)
{
    __shared__ float r1[1024], r2[1024], r3[1024];
    const int tid = threadIdx.x;
    float s = 0.f, s2 = 0.f, cnt = 0.f;
    for (int i = tid; i < TT; i += 1024) { const float v = surprise[i]; s += v; s2 += v * v; }
    for (int i = tid; i < NN; i += 1024) cnt += (sl0[i] > 0.0f) ? 1.0f : 0.0f;
    r1[tid] = s; r2[tid] = s2; r3[tid] = cnt; __syncthreads();
    for (int st = 512; st > 0; st >>= 1) {
        if (tid < st) { r1[tid] += r1[tid + st]; r2[tid] += r2[tid + st]; r3[tid] += r3[tid + st]; }
        __syncthreads();
    }
    if (tid == 0) {
        const float mean = r1[0] / (float)TT;
        float var = (r2[0] - (float)TT * mean * mean) / (float)(TT - 1);
        const float sd = fmaxf(sqrtf(fmaxf(var, 0.0f)), 1e-6f);
        const float util = r3[0] / (float)NN;
        *thr = (mean + 1.0f * sd) * (1.0f + util * 0.1f);
    }
}

// ---------------------------------------------------------------------------
// Row L2 norms (clamped to eps) — one wave32 per row.
// ---------------------------------------------------------------------------
__global__ __launch_bounds__(256)
void rownorm_kernel(const float* __restrict__ Ain, float* __restrict__ out, int R)
{
    const int row  = blockIdx.x * 8 + (threadIdx.x >> 5);
    const int lane = threadIdx.x & 31;
    if (row >= R) return;
    float acc = 0.0f;
    for (int d = lane; d < DD; d += 32) { const float v = Ain[(size_t)row * DD + d]; acc += v * v; }
    for (int o = 16; o > 0; o >>= 1) acc += __shfl_down(acc, o);
    if (lane == 0) out[row] = fmaxf(sqrtf(acc), 1e-8f);
}

// ---------------------------------------------------------------------------
// Sequential slot-contention scan. One persistent workgroup (32 wave32s).
// Slot metadata + similarity buffers live in LDS (~148 KB of the 320 KB WGP LDS).
// base_sim holds <K_snapshot[n], k_t>; only "dirty" (re-written) slots need a
// fresh dot product, done wave-parallel (one wave32 per dirty slot).
// ---------------------------------------------------------------------------
__global__ __launch_bounds__(1024)
void scan_kernel(const float* __restrict__ base_sim,  // [TT, NN] raw dots
                 const float* __restrict__ k_tok,     // [TT, DD]
                 const float* __restrict__ v_tok,     // [TT, DD]
                 const float* __restrict__ surprise,  // [TT]
                 const float* __restrict__ thr_p,     // [1]
                 const float* __restrict__ ktnorm,    // [TT]
                 const float* __restrict__ sl0, const float* __restrict__ ac0,
                 const float* __restrict__ sa0,
                 float* __restrict__ Kmem, float* __restrict__ Vmem, // [NN, DD]
                 float* __restrict__ Knorm,                           // [NN]
                 float* __restrict__ sl_out)                          // [NN]
{
    extern __shared__ char smem_raw[];
    float* s_sl  = (float*)smem_raw;          // NN
    float* s_ac  = s_sl  + NN;                // NN
    float* s_sa  = s_ac  + NN;                // NN
    float* s_sim = s_sa  + NN;                // NN
    float* s_kt  = s_sim + NN;                // DD
    float* r_v   = s_kt  + DD;                // 1024
    int*   r_i   = (int*)(r_v + 1024);        // 1024
    int*   s_meta = r_i + 1024;               // [0]=ndirty
    unsigned* s_dbit = (unsigned*)(s_meta + 8);          // NN/32 words
    unsigned short* s_dl = (unsigned short*)(s_dbit + NN / 32); // TT entries

    const int tid  = threadIdx.x;
    const int lane = tid & 31;
    const int wid  = tid >> 5;    // 0..31
    const int NW   = blockDim.x >> 5;

    for (int n = tid; n < NN; n += blockDim.x) {
        s_sl[n] = sl0[n]; s_ac[n] = ac0[n]; s_sa[n] = sa0[n];
    }
    for (int w = tid; w < NN / 32; w += blockDim.x) s_dbit[w] = 0u;
    if (tid == 0) s_meta[0] = 0;
    __syncthreads();
    const float thr = *thr_p;

    for (int t = 0; t < TT; ++t) {
        const float sval = surprise[t];
        if (!(sval > thr)) continue;          // uniform: nothing changes this step
        const float ktn = ktnorm[t];

        for (int d = tid; d < DD; d += blockDim.x) s_kt[d] = k_tok[(size_t)t * DD + d];
        __syncthreads();

        // Fill cosine sims from the snapshot GEMM.
        const float* br = base_sim + (size_t)t * NN;
        for (int n = tid; n < NN; n += blockDim.x)
            s_sim[n] = br[n] / (Knorm[n] * ktn);
        __syncthreads();

        // Re-dot the dirty slots (one wave per slot).
        const int nd = s_meta[0];
        for (int di = wid; di < nd; di += NW) {
            const int n = s_dl[di];
            const float* kr = Kmem + (size_t)n * DD;
            float acc = 0.0f;
            for (int d = lane; d < DD; d += 32) acc += kr[d] * s_kt[d];
            for (int o = 16; o > 0; o >>= 1) acc += __shfl_down(acc, o);
            if (lane == 0) s_sim[n] = acc / (Knorm[n] * ktn);
        }
        __syncthreads();

        // argmax(sim)
        float bmax = -__builtin_inff(); int bmaxi = 0;
        for (int n = tid; n < NN; n += blockDim.x) {
            const float v = s_sim[n];
            if (v > bmax) { bmax = v; bmaxi = n; }
        }
        r_v[tid] = bmax; r_i[tid] = bmaxi; __syncthreads();
        for (int st = 512; st > 0; st >>= 1) {
            if (tid < st && r_v[tid + st] > r_v[tid]) { r_v[tid] = r_v[tid + st]; r_i[tid] = r_i[tid + st]; }
            __syncthreads();
        }
        const float simmax = r_v[0];
        const int   amax   = r_i[0];
        __syncthreads();

        // argmin(importance), importance = 0.5*sl + 0.3*log1p(ac) - 0.001*sa
        float bmin = __builtin_inff(); int bmini = 0;
        for (int n = tid; n < NN; n += blockDim.x) {
            const float imp = 0.5f * s_sl[n] + 0.3f * log1pf(s_ac[n]) - 0.001f * s_sa[n];
            if (imp < bmin) { bmin = imp; bmini = n; }
        }
        r_v[tid] = bmin; r_i[tid] = bmini; __syncthreads();
        for (int st = 512; st > 0; st >>= 1) {
            if (tid < st && r_v[tid + st] < r_v[tid]) { r_v[tid] = r_v[tid + st]; r_i[tid] = r_i[tid + st]; }
            __syncthreads();
        }
        const int amin = r_i[0];
        __syncthreads();

        const bool consolidate = simmax > 0.85f;
        const int  slot = consolidate ? amax : amin;
        const bool do_update = consolidate && (sval > s_sl[slot]);
        const bool do_write  = !consolidate;

        if (do_update || do_write) {
            const float a  = do_update ? 0.3f : 1.0f;   // new = (1-a)*old + a*tok
            const float om = 1.0f - a;
            float sq = 0.0f;
            for (int d = tid; d < DD; d += blockDim.x) {
                const size_t idx = (size_t)slot * DD + d;
                const float nk = om * Kmem[idx] + a * s_kt[d];
                Kmem[idx] = nk;
                sq += nk * nk;
                const float nv = om * Vmem[idx] + a * v_tok[(size_t)t * DD + d];
                Vmem[idx] = nv;
            }
            r_v[tid] = sq; __syncthreads();
            for (int st = 512; st > 0; st >>= 1) {
                if (tid < st) r_v[tid] += r_v[tid + st];
                __syncthreads();
            }
            if (tid == 0) {
                Knorm[slot] = fmaxf(sqrtf(r_v[0]), 1e-8f);
                s_sl[slot]  = sval;
                if (do_write) s_ac[slot] = 0.0f;
                const unsigned w = (unsigned)slot >> 5, b = 1u << (slot & 31);
                if (!(s_dbit[w] & b)) { s_dbit[w] |= b; const int di = s_meta[0]++; s_dl[di] = (unsigned short)slot; }
            }
        }
        if (tid == 0) s_sa[slot] = 0.0f;  // write==true for any token reaching here
        __syncthreads();
    }

    for (int n = tid; n < NN; n += blockDim.x) sl_out[n] = s_sl[n];
}

// ---------------------------------------------------------------------------
// In-place row softmax over NN slots (handles -inf mask).
// ---------------------------------------------------------------------------
__global__ __launch_bounds__(256)
void softmax_kernel(float* __restrict__ logits)
{
    const int row = blockIdx.x;
    float* p = logits + (size_t)row * NN;
    __shared__ float red[256];
    float m = -__builtin_inff();
    for (int n = threadIdx.x; n < NN; n += 256) m = fmaxf(m, p[n]);
    red[threadIdx.x] = m; __syncthreads();
    for (int s = 128; s > 0; s >>= 1) {
        if (threadIdx.x < s) red[threadIdx.x] = fmaxf(red[threadIdx.x], red[threadIdx.x + s]);
        __syncthreads();
    }
    m = red[0]; __syncthreads();
    float sum = 0.0f;
    for (int n = threadIdx.x; n < NN; n += 256) { const float e = __expf(p[n] - m); p[n] = e; sum += e; }
    red[threadIdx.x] = sum; __syncthreads();
    for (int s = 128; s > 0; s >>= 1) {
        if (threadIdx.x < s) red[threadIdx.x] += red[threadIdx.x + s];
        __syncthreads();
    }
    const float inv = 1.0f / red[0];
    __syncthreads();
    for (int n = threadIdx.x; n < NN; n += 256) p[n] *= inv;
}

// ---------------------------------------------------------------------------
extern "C" void kernel_launch(void* const* d_in, const int* in_sizes, int n_in,
                              void* d_out, int out_size, void* d_ws, size_t ws_size,
                              hipStream_t stream)
{
    const float* x    = (const float*)d_in[0];
    const float* key0 = (const float*)d_in[1];
    const float* val0 = (const float*)d_in[2];
    const float* sl0  = (const float*)d_in[3];
    const float* ac0  = (const float*)d_in[4];
    const float* sa0  = (const float*)d_in[5];
    const float* Wc  = (const float*)d_in[6];  const float* bc  = (const float*)d_in[7];
    const float* Wp1 = (const float*)d_in[8];  const float* bp1 = (const float*)d_in[9];
    const float* Wp2 = (const float*)d_in[10]; const float* bp2 = (const float*)d_in[11];
    const float* Wk  = (const float*)d_in[12]; const float* bk  = (const float*)d_in[13];
    const float* Wv  = (const float*)d_in[14]; const float* bv  = (const float*)d_in[15];
    const float* Wq  = (const float*)d_in[16]; const float* bq  = (const float*)d_in[17];

    // Workspace layout (~176 MB, all L2-resident on MI455X's 192 MB L2).
    float* ws = (float*)d_ws;
    size_t o = 0;
    auto alloc = [&](size_t elems) { float* p = ws + o; o += elems; return p; };
    float* ctx    = alloc((size_t)TT * DD);
    float* h      = alloc((size_t)TT * DD);
    float* pred   = alloc((size_t)TT * DD);   // also holds ctx_in before step 3
    float* ktok   = alloc((size_t)TT * DD);
    float* vtok   = alloc((size_t)TT * DD);
    float* q      = alloc((size_t)TT * DD);
    float* Kmem   = alloc((size_t)NN * DD);
    float* Vmem   = alloc((size_t)NN * DD);
    float* logits = alloc((size_t)TT * NN);   // doubles as base_sim (dead after scan)
    float* surpr  = alloc(TT);
    float* Knorm  = alloc(NN);
    float* ktnorm = alloc(TT);
    float* thr    = alloc(1);
    float* slout  = alloc(NN);
    (void)ws_size; (void)in_sizes; (void)n_in; (void)out_size;

    const dim3 blk(256);
    auto gemm = [&](const float* A, const float* Bm, const float* bias, const float* mask,
                    float* Y, int M, int K, int N_, int bT, int act, float scale) {
        dim3 grid(M / 64, N_ / 128);
        hipLaunchKernelGGL(wmma_gemm_kernel, grid, blk, 0, stream,
                           A, Bm, bias, mask, Y, M, K, N_, bT, act, scale);
    };

    // --- surprise pipeline ---
    float* ctxin = pred;   // pred buffer is free until step 3
    hipLaunchKernelGGL(shift_tokens_kernel, dim3(TT), blk, 0, stream, x, ctxin);
    gemm(ctxin, Wc,  bc,  nullptr, ctx,  TT, DD, DD, 0, 0, 1.0f);
    gemm(ctx,   Wp1, bp1, nullptr, h,    TT, DD, DD, 0, 1, 1.0f);   // exact GELU
    gemm(h,     Wp2, bp2, nullptr, pred, TT, DD, DD, 0, 0, 1.0f);
    hipLaunchKernelGGL(surprise_kernel, dim3(TT), dim3(256), 0, stream, x, pred, surpr);
    hipLaunchKernelGGL(threshold_kernel, dim3(1), dim3(1024), 0, stream, surpr, sl0, thr);

    // --- projections ---
    gemm(x, Wk, bk, nullptr, ktok, TT, DD, DD, 0, 0, 1.0f);
    gemm(x, Wv, bv, nullptr, vtok, TT, DD, DD, 0, 0, 1.0f);
    gemm(x, Wq, bq, nullptr, q,    TT, DD, DD, 0, 0, 1.0f);

    // --- mutable memory copies + norms + similarity snapshot GEMM ---
    (void)hipMemcpyAsync(Kmem, key0, (size_t)NN * DD * sizeof(float), hipMemcpyDeviceToDevice, stream);
    (void)hipMemcpyAsync(Vmem, val0, (size_t)NN * DD * sizeof(float), hipMemcpyDeviceToDevice, stream);
    hipLaunchKernelGGL(rownorm_kernel, dim3(NN / 8), dim3(256), 0, stream, Kmem, Knorm, NN);
    hipLaunchKernelGGL(rownorm_kernel, dim3(TT / 8), dim3(256), 0, stream, ktok, ktnorm, TT);
    gemm(ktok, Kmem, nullptr, nullptr, logits /*base_sim*/, TT, DD, NN, 1, 0, 1.0f);

    // --- sequential slot-contention scan (single WGP, metadata in LDS) ---
    const size_t scan_lds = (size_t)(4 * NN + DD + 1024) * sizeof(float)  // sl/ac/sa/sim/kt/r_v
                          + 1024 * sizeof(int) + 8 * sizeof(int)          // r_i + meta
                          + (NN / 32) * sizeof(unsigned)                  // dirty bitmask
                          + TT * sizeof(unsigned short);                  // dirty list
    hipLaunchKernelGGL(scan_kernel, dim3(1), dim3(1024), scan_lds, stream,
                       logits, ktok, vtok, surpr, thr, ktnorm, sl0, ac0, sa0,
                       Kmem, Vmem, Knorm, slout);

    // --- attention read: masked logits -> softmax -> attn @ V ---
    gemm(q, Kmem, nullptr, slout, logits, TT, DD, NN, 1, 0, 0.03125f /*1/sqrt(1024)*/);
    hipLaunchKernelGGL(softmax_kernel, dim3(TT), dim3(256), 0, stream, logits);
    gemm(logits, Vmem, nullptr, nullptr, (float*)d_out, TT, NN, DD, 0, 0, 1.0f);
}
